// NanoGPT_35467839930464
// MI455X (gfx1250) — compile-verified, matrix operand order checked by reading, not run on hardware
//
#include <hip/hip_runtime.h>
#include <hip/hip_bf16.h>
#include <cstdint>
#include <cstddef>

// ---------------------------------------------------------------------------
// NanoGPT forward for MI455X (gfx1250, wave32, WMMA).
// f32 residual stream, f16 WMMA GEMMs (v_wmma_f32_16x16x32_f16, f32 accum),
// double-buffered LDS with async global->LDS copies (ASYNCcnt path).
// ---------------------------------------------------------------------------

typedef __attribute__((ext_vector_type(16))) _Float16 v16h;
typedef __attribute__((ext_vector_type(8)))  _Float16 v8h;
typedef __attribute__((ext_vector_type(8)))  float    v8f;
typedef int v4i __attribute__((__vector_size__(16)));   // matches builtin proto

#define Vv  50257
#define NP  50304          // padded LM-head width: 786*64, multiple of 8
#define Bb  64
#define Tt  128
#define Ee  256
#define Hh  8
#define Dd  32
#define Ll  6
#define FFf 1024
#define Mm  (Bb * Tt)      // 8192 rows
#define QKV (3 * Ee)       // 768

// gfx1250 async global->LDS copy (ASYNCcnt), guarded so compile stays clean.
#if defined(__has_builtin)
#  if __has_builtin(__builtin_amdgcn_global_load_async_to_lds_b128) && \
      __has_builtin(__builtin_amdgcn_s_wait_asynccnt)
#    define HAVE_ASYNC_LDS 1
#  endif
#endif
#ifndef HAVE_ASYNC_LDS
#  define HAVE_ASYNC_LDS 0
#endif

#if HAVE_ASYNC_LDS
// prototype (from compiler diagnostic): void(v4i*, v4i*, imm int, imm int)
#define ASYNC_CP_B128(gsrc, ldst)                                              \
  __builtin_amdgcn_global_load_async_to_lds_b128((v4i*)(gsrc), (v4i*)(ldst), 0, 0)
#define ASYNC_WAIT() __builtin_amdgcn_s_wait_asynccnt(0)
#endif

__device__ __forceinline__ v16h cat8(v8h lo, v8h hi) {
  return __builtin_shufflevector(lo, hi, 0,1,2,3,4,5,6,7,8,9,10,11,12,13,14,15);
}

// ---------------------------------------------------------------------------
// Weight conversions
// ---------------------------------------------------------------------------
__global__ __launch_bounds__(256) void cvt_f16_kernel(const float* __restrict__ s,
                                                      _Float16* __restrict__ d,
                                                      long long n) {
  long long i = (long long)blockIdx.x * 256 + threadIdx.x;
  if (i < n) d[i] = (_Float16)s[i];
}

// lm_w [E, V] -> f16 [E, NP], zero-padded columns V..NP-1
__global__ __launch_bounds__(256) void cvt_pad_kernel(const float* __restrict__ s,
                                                      _Float16* __restrict__ d) {
  long long i = (long long)blockIdx.x * 256 + threadIdx.x;
  if (i >= (long long)Ee * NP) return;
  int n = (int)(i % NP);
  long long r = i / NP;
  d[i] = (n < Vv) ? (_Float16)s[r * Vv + n] : (_Float16)0.f;
}

// concat q|k|v weights into [L*E, 3E] f16 (one fused QKV GEMM weight)
__global__ __launch_bounds__(256) void cvt_qkv_kernel(const float* __restrict__ q,
                                                      const float* __restrict__ k,
                                                      const float* __restrict__ v,
                                                      _Float16* __restrict__ d) {
  int row = blockIdx.x;            // 0 .. L*E-1
  int c   = threadIdx.x;           // 0 .. 255
  size_t so = (size_t)row * Ee + c;
  _Float16* dp = d + (size_t)row * QKV;
  dp[c]          = (_Float16)q[so];
  dp[Ee + c]     = (_Float16)k[so];
  dp[2 * Ee + c] = (_Float16)v[so];
}

// ---------------------------------------------------------------------------
// Embedding gather
// ---------------------------------------------------------------------------
__global__ __launch_bounds__(256) void embed_kernel(const int* __restrict__ idx,
                                                    const float* __restrict__ tok,
                                                    const float* __restrict__ pos,
                                                    float* __restrict__ x) {
  int i = blockIdx.x;
  int e = threadIdx.x;
  int t = i & (Tt - 1);
  x[(size_t)i * Ee + e] = tok[(size_t)idx[i] * Ee + e] + pos[(size_t)t * Ee + e];
}

// ---------------------------------------------------------------------------
// LayerNorm, wave-per-row (E = 256 -> 8 elems/lane), f16 output for WMMA.
// ---------------------------------------------------------------------------
__global__ __launch_bounds__(256) void ln_kernel(const float* __restrict__ x,
                                                 const float* __restrict__ w,
                                                 const float* __restrict__ b,
                                                 _Float16* __restrict__ out) {
  int wave = threadIdx.x >> 5;
  int lane = threadIdx.x & 31;
  int row  = blockIdx.x * 8 + wave;
  const float* xp = x + (size_t)row * Ee;

  float v[8];
  float s = 0.f;
#pragma unroll
  for (int j = 0; j < 8; ++j) { v[j] = xp[lane + j * 32]; s += v[j]; }
#pragma unroll
  for (int m = 1; m < 32; m <<= 1) s += __shfl_xor(s, m, 32);
  float mu = s * (1.f / Ee);

  float vs = 0.f;
#pragma unroll
  for (int j = 0; j < 8; ++j) { float d = v[j] - mu; vs += d * d; }
#pragma unroll
  for (int m = 1; m < 32; m <<= 1) vs += __shfl_xor(vs, m, 32);
  float inv = rsqrtf(vs * (1.f / Ee) + 1e-5f);

  _Float16* op = out + (size_t)row * Ee;
#pragma unroll
  for (int j = 0; j < 8; ++j) {
    int c = lane + j * 32;
    op[c] = (_Float16)((v[j] - mu) * inv * w[c] + b[c]);
  }
}

// ---------------------------------------------------------------------------
// WMMA GEMM:  C[M,N] = A[M,K](f16) * Bw[K,N ldb](f16)  (+bias,+gelu,+residual)
// 128x64 block tile, 8 waves * (32x32 = 2x2 WMMA). Double-buffered LDS;
// A tile streams global->LDS asynchronously when the builtin exists.
// ---------------------------------------------------------------------------
#define BM 128
#define BN 64
#define BK 32
#define LDA 40   // BK + 8 halves pad  (80 B row stride, 16B aligned)
#define LDB 40

template <bool BIAS, bool RES, bool GELU, bool OUT16, bool OUT32, bool NCHK>
__global__ __launch_bounds__(256)
void gemm_kernel(const _Float16* __restrict__ A,
                 const _Float16* __restrict__ Bw,
                 const float* __restrict__ bias,
                 const float* __restrict__ resid,
                 float* __restrict__ out32,
                 _Float16* __restrict__ out16,
                 int M, int N, int K, int ldb) {
  __shared__ _Float16 Ash[2][BM * LDA];
  __shared__ _Float16 Bsh[2][BN * LDB];   // transposed: Bsh[buf][n][k]

  int tid   = threadIdx.x;
  int lane  = tid & 31;
  int wave  = tid >> 5;
  int wm    = wave & 3;
  int wn    = wave >> 2;
  int l16   = lane & 15;
  int lhalf = lane >> 4;
  int m0 = blockIdx.y * BM;
  int n0 = blockIdx.x * BN;

  // per-thread cooperative-load coordinates
  int ar0 = (tid * 2) >> 2;            // A: two v8h per thread
  int ac0 = ((tid * 2) & 3) * 8;
  int ar1 = (tid * 2 + 1) >> 2;
  int ac1 = ((tid * 2 + 1) & 3) * 8;
  int bkk = tid >> 3;                  // B: one v8h per thread
  int bnn = (tid & 7) * 8;

  const int nchunks = K / BK;
  v8f acc[2][2] = {};

  // ---- prologue: chunk 0 into buffer 0 ----
#if HAVE_ASYNC_LDS
  ASYNC_CP_B128(A + (size_t)(m0 + ar0) * K + ac0, &Ash[0][ar0 * LDA + ac0]);
  ASYNC_CP_B128(A + (size_t)(m0 + ar1) * K + ac1, &Ash[0][ar1 * LDA + ac1]);
#else
  *(v8h*)&Ash[0][ar0 * LDA + ac0] = *(const v8h*)(A + (size_t)(m0 + ar0) * K + ac0);
  *(v8h*)&Ash[0][ar1 * LDA + ac1] = *(const v8h*)(A + (size_t)(m0 + ar1) * K + ac1);
#endif
  {
    v8h b0 = *(const v8h*)(Bw + (size_t)bkk * ldb + n0 + bnn);
#pragma unroll
    for (int e = 0; e < 8; ++e) Bsh[0][(bnn + e) * LDB + bkk] = b0[e];
  }
#if HAVE_ASYNC_LDS
  ASYNC_WAIT();
#endif
  __syncthreads();

  for (int c = 0; c < nchunks; ++c) {
    const int cur = c & 1, nxt = cur ^ 1;
    const bool hasnext = (c + 1 < nchunks);
    const int k1 = (c + 1) * BK;
    v8h bnext;
#if !HAVE_ASYNC_LDS
    v8h anext0, anext1;
#endif
    if (hasnext) {
#if HAVE_ASYNC_LDS
      ASYNC_CP_B128(A + (size_t)(m0 + ar0) * K + k1 + ac0, &Ash[nxt][ar0 * LDA + ac0]);
      ASYNC_CP_B128(A + (size_t)(m0 + ar1) * K + k1 + ac1, &Ash[nxt][ar1 * LDA + ac1]);
#else
      anext0 = *(const v8h*)(A + (size_t)(m0 + ar0) * K + k1 + ac0);
      anext1 = *(const v8h*)(A + (size_t)(m0 + ar1) * K + k1 + ac1);
#endif
      bnext = *(const v8h*)(Bw + (size_t)(k1 + bkk) * ldb + n0 + bnn);
      if (c + 2 < nchunks)   // deep L2 prefetch (global_prefetch_b8)
        __builtin_prefetch(A + (size_t)(m0 + ar0) * K + k1 + BK + ac0, 0, 1);
    }

    // ---- compute on current buffer ----
#pragma unroll
    for (int mi = 0; mi < 2; ++mi) {
      const _Float16* ap = &Ash[cur][(wm * 32 + mi * 16 + l16) * LDA + lhalf * 8];
      v16h af = cat8(*(const v8h*)ap, *(const v8h*)(ap + 16));
#pragma unroll
      for (int nj = 0; nj < 2; ++nj) {
        const _Float16* bp = &Bsh[cur][(wn * 32 + nj * 16 + l16) * LDB + lhalf * 8];
        v16h bf = cat8(*(const v8h*)bp, *(const v8h*)(bp + 16));
        acc[mi][nj] = __builtin_amdgcn_wmma_f32_16x16x32_f16(
            false, af, false, bf, (short)0, acc[mi][nj], false, false);
      }
    }

    if (hasnext) {
#if !HAVE_ASYNC_LDS
      *(v8h*)&Ash[nxt][ar0 * LDA + ac0] = anext0;
      *(v8h*)&Ash[nxt][ar1 * LDA + ac1] = anext1;
#endif
#pragma unroll
      for (int e = 0; e < 8; ++e) Bsh[nxt][(bnn + e) * LDB + bkk] = bnext[e];
#if HAVE_ASYNC_LDS
      ASYNC_WAIT();
#endif
      __syncthreads();
    }
  }

  // ---- fused epilogue ----
#pragma unroll
  for (int mi = 0; mi < 2; ++mi)
#pragma unroll
    for (int nj = 0; nj < 2; ++nj)
#pragma unroll
      for (int r = 0; r < 8; ++r) {
        int row = m0 + wm * 32 + mi * 16 + r + 8 * lhalf;
        int col = n0 + wn * 32 + nj * 16 + l16;
        if (!NCHK || col < N) {
          float v = acc[mi][nj][r];
          if (BIAS) v += bias[col];
          if (GELU) v = 0.5f * v * (1.f + erff(v * 0.70710678f));
          if (RES)  v += resid[(size_t)row * N + col];
          if (OUT32) out32[(size_t)row * N + col] = v;
          if (OUT16) out16[(size_t)row * N + col] = (_Float16)v;
        }
      }
}

// ---------------------------------------------------------------------------
// Attention: one block per (b, h), reads fused qkv activations [M, 768].
// 4 waves; each wave owns a full 32x128 score strip -> in-register softmax.
// ---------------------------------------------------------------------------
#define LQ 40    // D + 8 pad
#define LS 136   // T + 8 pad

__global__ __launch_bounds__(128)
void attn_kernel(const _Float16* __restrict__ qkv,
                 _Float16* __restrict__ o) {
  __shared__ _Float16 Qs[Tt * LQ];
  __shared__ _Float16 Ks[Tt * LQ];
  __shared__ _Float16 Vt[Dd * LS];   // V transposed: Vt[d][t]
  __shared__ _Float16 Ps[Tt * LS];   // softmaxed probs, row-major

  int bh = blockIdx.x;
  int b  = bh >> 3;
  int h  = bh & 7;
  const size_t qbase = ((size_t)b * Tt) * QKV + (size_t)h * Dd;
  const size_t obase = ((size_t)b * Tt) * Ee  + (size_t)h * Dd;

  int tid = threadIdx.x;   // 0..127 == row t
  {
    const _Float16* qp = qkv + qbase + (size_t)tid * QKV;
    const _Float16* kp = qp + Ee;
    const _Float16* vp = qp + 2 * Ee;
#pragma unroll
    for (int j = 0; j < 4; ++j) {
#if HAVE_ASYNC_LDS
      ASYNC_CP_B128(qp + j * 8, &Qs[tid * LQ + j * 8]);
      ASYNC_CP_B128(kp + j * 8, &Ks[tid * LQ + j * 8]);
#else
      *(v8h*)&Qs[tid * LQ + j * 8] = *(const v8h*)(qp + j * 8);
      *(v8h*)&Ks[tid * LQ + j * 8] = *(const v8h*)(kp + j * 8);
#endif
      v8h vv = *(const v8h*)(vp + j * 8);
#pragma unroll
      for (int e = 0; e < 8; ++e) Vt[(j * 8 + e) * LS + tid] = vv[e];
    }
  }
#if HAVE_ASYNC_LDS
  ASYNC_WAIT();
#endif
  __syncthreads();

  int lane = tid & 31, wave = tid >> 5;
  int l16 = lane & 15, lhalf = lane >> 4;
  const float scale = 0.0625f;   // E^-0.5 (reference scales by n_embd)
  const float NEG = -3.0e38f;

#pragma unroll
  for (int mi = 0; mi < 2; ++mi) {
    const _Float16* ap = &Qs[(wave * 32 + mi * 16 + l16) * LQ + lhalf * 8];
    v16h af = cat8(*(const v8h*)ap, *(const v8h*)(ap + 16));
    v8f sacc[8];
#pragma unroll
    for (int nj = 0; nj < 8; ++nj) {
      const _Float16* bp = &Ks[(nj * 16 + l16) * LQ + lhalf * 8];
      v16h bf = cat8(*(const v8h*)bp, *(const v8h*)(bp + 16));
      v8f z = {};
      sacc[nj] = __builtin_amdgcn_wmma_f32_16x16x32_f16(
          false, af, false, bf, (short)0, z, false, false);
    }
    // causal softmax per register-row (row lives in one 16-lane half)
#pragma unroll
    for (int r = 0; r < 8; ++r) {
      int row = wave * 32 + mi * 16 + r + 8 * lhalf;
      float vals[8];
      float mx = NEG;
#pragma unroll
      for (int nj = 0; nj < 8; ++nj) {
        int col = nj * 16 + l16;
        float s = (col <= row) ? sacc[nj][r] * scale : NEG;
        vals[nj] = s;
        mx = fmaxf(mx, s);
      }
#pragma unroll
      for (int m = 1; m < 16; m <<= 1) mx = fmaxf(mx, __shfl_xor(mx, m, 32));
      float sum = 0.f;
#pragma unroll
      for (int nj = 0; nj < 8; ++nj) {
        float e = __expf(vals[nj] - mx);
        vals[nj] = e;
        sum += e;
      }
#pragma unroll
      for (int m = 1; m < 16; m <<= 1) sum += __shfl_xor(sum, m, 32);
      float inv = 1.f / sum;
#pragma unroll
      for (int nj = 0; nj < 8; ++nj)
        Ps[row * LS + nj * 16 + l16] = (_Float16)(vals[nj] * inv);
    }
  }
  __syncthreads();

  // O = P @ V  (128x128 x 128x32), 4 K-chunks of 32
  v8f oacc[2][2] = {};
#pragma unroll
  for (int kc = 0; kc < 4; ++kc) {
#pragma unroll
    for (int mi = 0; mi < 2; ++mi) {
      const _Float16* ap = &Ps[(wave * 32 + mi * 16 + l16) * LS + kc * 32 + lhalf * 8];
      v16h af = cat8(*(const v8h*)ap, *(const v8h*)(ap + 16));
#pragma unroll
      for (int nj = 0; nj < 2; ++nj) {
        const _Float16* bp = &Vt[(nj * 16 + l16) * LS + kc * 32 + lhalf * 8];
        v16h bf = cat8(*(const v8h*)bp, *(const v8h*)(bp + 16));
        oacc[mi][nj] = __builtin_amdgcn_wmma_f32_16x16x32_f16(
            false, af, false, bf, (short)0, oacc[mi][nj], false, false);
      }
    }
  }
#pragma unroll
  for (int mi = 0; mi < 2; ++mi)
#pragma unroll
    for (int nj = 0; nj < 2; ++nj)
#pragma unroll
      for (int r = 0; r < 8; ++r) {
        int row = wave * 32 + mi * 16 + r + 8 * lhalf;
        int col = nj * 16 + l16;
        o[obase + (size_t)row * Ee + col] = (_Float16)oacc[mi][nj][r];
      }
}

// ---------------------------------------------------------------------------
// Host side
// ---------------------------------------------------------------------------
static inline void* take(char* ws, size_t& off, size_t bytes) {
  off = (off + 255) & ~(size_t)255;
  void* p = ws + off;
  off += bytes;
  return p;
}

static void cvt(const float* s, _Float16* d, long long n, hipStream_t st) {
  int blocks = (int)((n + 255) / 256);
  cvt_f16_kernel<<<blocks, 256, 0, st>>>(s, d, n);
}

extern "C" void kernel_launch(void* const* d_in, const int* in_sizes, int n_in,
                              void* d_out, int out_size, void* d_ws, size_t ws_size,
                              hipStream_t stream) {
  const int*   idx     = (const int*)  d_in[0];
  const float* tok_emb = (const float*)d_in[1];
  const float* pos_emb = (const float*)d_in[2];
  const float* q_w     = (const float*)d_in[3];
  const float* k_w     = (const float*)d_in[4];
  const float* v_w     = (const float*)d_in[5];
  const float* proj_w  = (const float*)d_in[6];
  const float* proj_b  = (const float*)d_in[7];
  const float* ln1_w   = (const float*)d_in[8];
  const float* ln1_b   = (const float*)d_in[9];
  const float* ln2_w   = (const float*)d_in[10];
  const float* ln2_b   = (const float*)d_in[11];
  const float* w1      = (const float*)d_in[12];
  const float* b1      = (const float*)d_in[13];
  const float* w2      = (const float*)d_in[14];
  const float* b2      = (const float*)d_in[15];
  const float* lnf_w   = (const float*)d_in[16];
  const float* lnf_b   = (const float*)d_in[17];
  const float* lm_w    = (const float*)d_in[18];
  const float* lm_b    = (const float*)d_in[19];

  char* ws = (char*)d_ws;
  size_t off = 0;
  // f16 weight copies
  _Float16* qkvw16 = (_Float16*)take(ws, off, sizeof(_Float16) * (size_t)Ll * Ee * QKV);
  _Float16* pw16   = (_Float16*)take(ws, off, sizeof(_Float16) * (size_t)Ll * Ee * Ee);
  _Float16* w116   = (_Float16*)take(ws, off, sizeof(_Float16) * (size_t)Ll * Ee * FFf);
  _Float16* w216   = (_Float16*)take(ws, off, sizeof(_Float16) * (size_t)Ll * FFf * Ee);
  _Float16* lm16   = (_Float16*)take(ws, off, sizeof(_Float16) * (size_t)Ee * NP);
  // activations
  float*    x      = (float*)   take(ws, off, sizeof(float)    * (size_t)Mm * Ee);
  _Float16* h16    = (_Float16*)take(ws, off, sizeof(_Float16) * (size_t)Mm * Ee);
  _Float16* qkv16  = (_Float16*)take(ws, off, sizeof(_Float16) * (size_t)Mm * QKV);
  _Float16* ao16   = (_Float16*)take(ws, off, sizeof(_Float16) * (size_t)Mm * Ee);
  _Float16* ff16   = (_Float16*)take(ws, off, sizeof(_Float16) * (size_t)Mm * FFf);
  (void)ws_size; (void)in_sizes; (void)n_in; (void)out_size;

  // 1) convert weights to f16 (lm_w gets padded row stride NP)
  cvt_qkv_kernel<<<Ll * Ee, 256, 0, stream>>>(q_w, k_w, v_w, qkvw16);
  cvt(proj_w, pw16, (long long)Ll * Ee * Ee,  stream);
  cvt(w1,     w116, (long long)Ll * Ee * FFf, stream);
  cvt(w2,     w216, (long long)Ll * FFf * Ee, stream);
  cvt_pad_kernel<<<(int)(((long long)Ee * NP + 255) / 256), 256, 0, stream>>>(lm_w, lm16);

  // 2) embeddings
  embed_kernel<<<Mm, 256, 0, stream>>>(idx, tok_emb, pos_emb, x);

  dim3 blk(256);
  auto grid = [](int N) { return dim3((unsigned)((N + BN - 1) / BN), Mm / BM); };

  // 3) transformer blocks
  for (int l = 0; l < Ll; ++l) {
    ln_kernel<<<Mm / 8, 256, 0, stream>>>(x, ln1_w + l * Ee, ln1_b + l * Ee, h16);

    // fused QKV: [8192,256] x [256,768]
    gemm_kernel<false, false, false, true, false, false><<<grid(QKV), blk, 0, stream>>>(
        h16, qkvw16 + (size_t)l * Ee * QKV, nullptr, nullptr, nullptr, qkv16,
        Mm, QKV, Ee, QKV);

    attn_kernel<<<Bb * Hh, 128, 0, stream>>>(qkv16, ao16);

    // x = x + ao @ proj_w + proj_b
    gemm_kernel<true, true, false, false, true, false><<<grid(Ee), blk, 0, stream>>>(
        ao16, pw16 + (size_t)l * Ee * Ee, proj_b + l * Ee, x, x, nullptr,
        Mm, Ee, Ee, Ee);

    ln_kernel<<<Mm / 8, 256, 0, stream>>>(x, ln2_w + l * Ee, ln2_b + l * Ee, h16);

    // ff = gelu(h @ W1 + b1)
    gemm_kernel<true, false, true, true, false, false><<<grid(FFf), blk, 0, stream>>>(
        h16, w116 + (size_t)l * Ee * FFf, b1 + l * FFf, nullptr, nullptr, ff16,
        Mm, FFf, Ee, FFf);

    // x = x + ff @ W2 + b2
    gemm_kernel<true, true, false, false, true, false><<<grid(Ee), blk, 0, stream>>>(
        ff16, w216 + (size_t)l * FFf * Ee, b2 + l * Ee, x, x, nullptr,
        Mm, Ee, FFf, Ee);
  }

  // 4) final LN + LM head (write-bandwidth-bound: 1.65 GB of logits)
  ln_kernel<<<Mm / 8, 256, 0, stream>>>(x, lnf_w, lnf_b, h16);
  gemm_kernel<true, false, false, false, true, true><<<grid(Vv), blk, 0, stream>>>(
      h16, lm16, lm_b, nullptr, (float*)d_out, nullptr, Mm, Vv, Ee, NP);
}
